// GELU302_22677427323037
// MI455X (gfx1250) — matrix operand order repaired and checked by reading, compiled, not run on hardware
//
#include <hip/hip_runtime.h>
#include <math.h>

typedef __attribute__((ext_vector_type(16))) _Float16 v16h;
typedef __attribute__((ext_vector_type(8)))  float    v8f;

#define D_FF 4096
#define EPSV 1e-4f
#define EPSD 1e-5f

// ---- workspace layout (float offsets) ----
#define WS_PKIN   0        // float2[4096]: {ema_mean, inv_std_in (sign = topk mask)}
#define WS_PKOUT  8192     // float2[4096]: {ema_out_mean, inv_std_out}
#define WS_EMAN   16384    // float[4096]: normalized ema_out_dir
#define WS_NOVEL  20480    // float[4096]: partial sums of |x - mean|
#define WS_SCAL   24576    // float[6]: tau, beta_up, beta_dn, gamma, beta_out, gamma_out
#define WS_GCOS   24608    // float[nrows]: per-row gate_cos

// Branchless transcendentals: single v_exp_f32 / v_rcp_f32, no EXEC ladders.
__device__ __forceinline__ float tanh_fast(float y) {
  // tanh(y) = 1 - 2/(exp(2y)+1); exp2 saturates cleanly at +-inf -> +-1.
  float t = __builtin_amdgcn_exp2f(y * 2.8853900817779268f);  // 2*log2(e)
  return 1.0f - 2.0f * __builtin_amdgcn_rcpf(t + 1.0f);
}
__device__ __forceinline__ float exp_fast(float v) {
  return __builtin_amdgcn_exp2f(v * 1.4426950408889634f);
}
__device__ __forceinline__ float gelu_f(float x) {
  const float c = 0.7978845608028654f;  // sqrt(2/pi)
  return 0.5f * x * (1.0f + tanh_fast(c * (x + 0.044715f * x * x * x)));
}
__device__ __forceinline__ float softplus_f(float v) { return log1pf(expf(v)); }
__device__ __forceinline__ float clampf(float v, float lo, float hi) {
  return fminf(fmaxf(v, lo), hi);
}

// ---------------- K0: per-channel constants + scalars ----------------
__global__ void __launch_bounds__(1024)
k0_prep(const float* __restrict__ log_tau, const float* __restrict__ log_bu,
        const float* __restrict__ log_bd, const float* __restrict__ log_g,
        const float* __restrict__ log_bo, const float* __restrict__ log_go,
        const float* __restrict__ ema_mean, const float* __restrict__ ema_sq,
        const float* __restrict__ ema_om, const float* __restrict__ ema_osq,
        const float* __restrict__ ema_dir, float* __restrict__ W) {
  __shared__ float red[1024];
  const int t = threadIdx.x;
  float loc = 0.f;
  for (int d = t; d < D_FF; d += 1024) { float v = ema_dir[d]; loc += v * v; }
  red[t] = loc;
  __syncthreads();
  for (int s = 512; s > 0; s >>= 1) {
    if (t < s) red[t] += red[t + s];
    __syncthreads();
  }
  const float inv_n = 1.0f / fmaxf(sqrtf(red[0]), 1e-12f);

  float2* pkin  = (float2*)(W + WS_PKIN);
  float2* pkout = (float2*)(W + WS_PKOUT);
  for (int d = t; d < D_FF; d += 1024) {
    float m   = ema_mean[d];
    float var = fmaxf(ema_sq[d] - m * m, EPSV);
    pkin[d]   = make_float2(m, 1.0f / (sqrtf(var) + EPSD));
    float mo   = ema_om[d];
    float varo = fmaxf(ema_osq[d] - mo * mo, EPSV);
    pkout[d]   = make_float2(mo, 1.0f / (sqrtf(varo) + EPSD));
    W[WS_EMAN + d]  = ema_dir[d] * inv_n;
    W[WS_NOVEL + d] = 0.f;
  }
  if (t == 0) {
    W[WS_SCAL + 0] = expf(log_tau[0]);
    W[WS_SCAL + 1] = softplus_f(log_bu[0]);
    W[WS_SCAL + 2] = softplus_f(log_bd[0]);
    W[WS_SCAL + 3] = softplus_f(log_g[0]);
    W[WS_SCAL + 4] = softplus_f(log_bo[0]);
    W[WS_SCAL + 5] = softplus_f(log_go[0]);
  }
}

// ---------------- K1: WMMA row dots + row norms + column |x-mu| partials ----
// Block = 256 threads = 8 waves; each wave owns 16 rows; WMMA computes
// dot(gelu(x_row), ema_n) for 16 rows at once with ema_n in B column 0.
__global__ void __launch_bounds__(256)
k1_rowstats(const float* __restrict__ x, const float* __restrict__ ema_mean,
            float* __restrict__ W) {
  __shared__ float sdot[8][16];
  const int tid  = threadIdx.x;
  const int lane = tid & 31;
  const int wave = tid >> 5;
  const int m    = lane & 15;   // M index (row within wave's tile)
  const int h    = lane >> 4;   // K-half selector (A layout lanes 16-31)
  const int row  = blockIdx.x * 128 + wave * 16 + m;
  const float* xr   = x + (size_t)row * D_FF;
  const float* eman = W + WS_EMAN;
  const float  tau  = W[WS_SCAL + 0];

  float sumsq = 0.f;
  v8f c;
#pragma unroll
  for (int i = 0; i < 8; i++) c[i] = 0.f;

  for (int kb = 0; kb < D_FF; kb += 32) {
    // A 16x32 f16 layout: lane (m,h) holds K in {kb+8h..kb+8h+7} U {kb+16+8h..+7}
    const float4* p = (const float4*)(xr + kb + h * 8);
    const float4* q = (const float4*)(xr + kb + 16 + h * 8);
    float4 a0 = p[0], a1 = p[1], a2 = q[0], a3 = q[1];
    float o[16];
    o[0]  = gelu_f(a0.x); o[1]  = gelu_f(a0.y); o[2]  = gelu_f(a0.z); o[3]  = gelu_f(a0.w);
    o[4]  = gelu_f(a1.x); o[5]  = gelu_f(a1.y); o[6]  = gelu_f(a1.z); o[7]  = gelu_f(a1.w);
    o[8]  = gelu_f(a2.x); o[9]  = gelu_f(a2.y); o[10] = gelu_f(a2.z); o[11] = gelu_f(a2.w);
    o[12] = gelu_f(a3.x); o[13] = gelu_f(a3.y); o[14] = gelu_f(a3.z); o[15] = gelu_f(a3.w);
    v16h A, B;
#pragma unroll
    for (int i = 0; i < 16; i++) {
      sumsq += o[i] * o[i];
      A[i] = (_Float16)o[i];
      B[i] = (_Float16)0.f;
    }
    // B 32x16 f16 layout: lane n==0 (lanes 0,16) carries ema_n in column 0,
    // K = 8h + i for i < 8, K = 16 + 8h + (i - 8) for i >= 8.
    if (m == 0) {
      const float* e = eman + kb + h * 8;
#pragma unroll
      for (int i = 0; i < 16; i++) B[i] = (_Float16)e[i < 8 ? i : i + 8];
    }
    c = __builtin_amdgcn_wmma_f32_16x16x32_f16(false, A, false, B, (short)0, c,
                                               false, false);
  }
  // ||out_row||^2: lanes m and m+16 hold disjoint K halves of row m.
  float ss = sumsq + __shfl_xor(sumsq, 16, 32);
  // D column 0: lane 0 has M=0..7 in c[0..7], lane 16 has M=8..15.
  if (lane == 0)
    for (int j = 0; j < 8; j++) sdot[wave][j] = c[j];
  if (lane == 16)
    for (int j = 0; j < 8; j++) sdot[wave][8 + j] = c[j];
  __syncthreads();
  if (lane < 16) {
    float dot = sdot[wave][m];
    float nrm = fmaxf(sqrtf(ss), 1e-12f);
    float cs  = clampf(dot / nrm, -1.f, 1.f);
    W[WS_GCOS + row] = exp_fast(-tau * cs);
  }
  __syncthreads();

  // Phase 2: column partials of |x - mean| over this block's 128 rows
  // (rows just read -> L2-hot). Register accumulators, one atomic per channel.
  float acc[16], mu[16];
#pragma unroll
  for (int j = 0; j < 16; j++) { acc[j] = 0.f; mu[j] = ema_mean[tid + 256 * j]; }
  const float* xb = x + (size_t)(blockIdx.x * 128) * D_FF;
  for (int r = 0; r < 128; r++) {
    const float* xr2 = xb + (size_t)r * D_FF;
#pragma unroll
    for (int j = 0; j < 16; j++) acc[j] += fabsf(xr2[tid + 256 * j] - mu[j]);
  }
#pragma unroll
  for (int j = 0; j < 16; j++) atomicAdd(&W[WS_NOVEL + tid + 256 * j], acc[j]);
}

// ---------------- K2: top-k threshold via LDS bitonic sort ----------------
__global__ void __launch_bounds__(1024)
k2_topk(const int* __restrict__ kptr, int nrows, float* __restrict__ W) {
  __shared__ float ssort[D_FF];
  __shared__ float sorig[D_FF];
  const int t = threadIdx.x;
  float2* pkin = (float2*)(W + WS_PKIN);
  const float invR = 1.0f / (float)nrows;
  for (int d = t; d < D_FF; d += 1024) {
    float sc = W[WS_NOVEL + d] * pkin[d].y * invR;  // mean|z_in| per channel
    ssort[d] = sc;
    sorig[d] = sc;
  }
  __syncthreads();
  for (int ksz = 2; ksz <= D_FF; ksz <<= 1) {
    for (int j = ksz >> 1; j > 0; j >>= 1) {
      for (int i = t; i < D_FF; i += 1024) {
        int ixj = i ^ j;
        if (ixj > i) {
          float a = ssort[i], b = ssort[ixj];
          bool up = ((i & ksz) == 0);  // descending
          if (up ? (a < b) : (a > b)) { ssort[i] = b; ssort[ixj] = a; }
        }
      }
      __syncthreads();
    }
  }
  int k = kptr[0];
  if (k < 1) k = 1;
  if (k > D_FF) k = D_FF;
  const float thr = ssort[k - 1];
  for (int d = t; d < D_FF; d += 1024) {
    float inv = fabsf(pkin[d].y);
    pkin[d].y = (sorig[d] >= thr) ? inv : -inv;  // fold mask into sign bit
  }
}

// ---------------- K3: apply pass (async-staged LDS tables) ----------------
__global__ void __launch_bounds__(256)
k3_apply(const float* __restrict__ x, const float* __restrict__ W,
         float* __restrict__ out) {
  __shared__ float lds[16384];  // 64KB: [0..8191]=pkin, [8192..16383]=pkout
  const int t = threadIdx.x;
  // Stage pkin+pkout (contiguous 64KB at W) via gfx1250 async global->LDS DMA.
  {
    const char* src = (const char*)W;  // WS_PKIN == 0
#pragma unroll
    for (int q = 0; q < 16; q++) {
      unsigned off = (unsigned)((q * 256 + t) * 16);
      unsigned long long ga = (unsigned long long)(src + off);
      asm volatile("global_load_async_to_lds_b128 %0, %1, off"
                   :: "v"(off), "v"(ga) : "memory");
    }
    asm volatile("s_wait_asynccnt 0" ::: "memory");
  }
  __syncthreads();
  const float2* lpin  = (const float2*)(lds);
  const float2* lpout = (const float2*)(lds + 8192);
  const float bu = W[WS_SCAL + 1], bd = W[WS_SCAL + 2], gg = W[WS_SCAL + 3];
  const float bo = W[WS_SCAL + 4], go = W[WS_SCAL + 5];
  const int rb = blockIdx.x * 8;
  for (int rr = 0; rr < 8; rr++) {
    const int row = rb + rr;
    const float gcos = W[WS_GCOS + row];
    const float4* xr = (const float4*)(x + (size_t)row * D_FF);
    float4* orow = (float4*)(out + (size_t)row * D_FF);
#pragma unroll
    for (int q2 = 0; q2 < 4; q2++) {
      const int i4 = t + 256 * q2;
      float4 xv = xr[i4];
      float4 r;
      float* xp = &xv.x;
      float* rp = &r.x;
#pragma unroll
      for (int e = 0; e < 4; e++) {
        const int d = i4 * 4 + e;
        float xe = xp[e];
        float o  = gelu_f(xe);
        float2 pi = lpin[d];
        float inv = fabsf(pi.y);
        float msk = pi.y > 0.f ? 1.f : 0.f;
        float zin = (xe - pi.x) * inv;
        float th1 = tanh_fast(gg * zin);
        float gin = clampf(1.f + bu * fmaxf(th1, 0.f) - bd * fmaxf(-th1, 0.f),
                           0.05f, 8.f);
        float2 po  = lpout[d];
        float zout = (o - po.x) * po.y;
        float gout = clampf(1.f + bo * tanh_fast(go * zout), 0.1f, 5.f);
        float g = gin * gout * gcos;
        rp[e] = o * (g * msk + (1.f - msk));
      }
      orow[i4] = r;
    }
  }
}

extern "C" void kernel_launch(void* const* d_in, const int* in_sizes, int n_in,
                              void* d_out, int out_size, void* d_ws,
                              size_t ws_size, hipStream_t stream) {
  const float* x        = (const float*)d_in[0];
  // d_in[1] logit_decay: unused in forward
  const float* log_tau  = (const float*)d_in[2];
  const float* log_bu   = (const float*)d_in[3];
  const float* log_bd   = (const float*)d_in[4];
  const float* log_g    = (const float*)d_in[5];
  const float* log_bo   = (const float*)d_in[6];
  const float* log_go   = (const float*)d_in[7];
  // d_in[8] log_temperature: softmax STE term is 0 in forward value
  const float* ema_mean = (const float*)d_in[9];
  const float* ema_sq   = (const float*)d_in[10];
  const float* ema_om   = (const float*)d_in[11];
  const float* ema_osq  = (const float*)d_in[12];
  const float* ema_dir  = (const float*)d_in[13];
  const int*   kptr     = (const int*)d_in[14];
  float* W   = (float*)d_ws;
  float* out = (float*)d_out;
  const int nrows = in_sizes[0] / D_FF;  // 8192

  k0_prep<<<1, 1024, 0, stream>>>(log_tau, log_bu, log_bd, log_g, log_bo,
                                  log_go, ema_mean, ema_sq, ema_om, ema_osq,
                                  ema_dir, W);
  k1_rowstats<<<nrows / 128, 256, 0, stream>>>(x, ema_mean, W);
  k2_topk<<<1, 1024, 0, stream>>>(kptr, nrows, W);
  k3_apply<<<nrows / 8, 256, 0, stream>>>(x, W, out);
}
